// ContextQueryAttn_83820581749122
// MI455X (gfx1250) — matrix operand order repaired and checked by reading, not compiled, not run on hardware
//
#include <hip/hip_runtime.h>
#include <math.h>

// Problem constants (from reference): B, LC, LQ, D
constexpr int NB  = 32;
constexpr int LC  = 2048;
constexpr int LQ  = 256;
constexpr int DD  = 256;

typedef __attribute__((ext_vector_type(16))) __bf16          v16bf;
typedef __attribute__((ext_vector_type(8)))  float           v8f;
typedef __attribute__((ext_vector_type(8)))  unsigned short  u16x8;
typedef __attribute__((ext_vector_type(16))) unsigned short  u16x16;

__device__ __forceinline__ unsigned short f2bf(float f) {
  unsigned int u = __builtin_bit_cast(unsigned int, f);
  u += 0x7FFFu + ((u >> 16) & 1u);          // round-to-nearest-even
  return (unsigned short)(u >> 16);
}

// ---------------------------------------------------------------------------
// prep: sc = context . wc  (per row), cwm = bf16(context * wm)
// one wave per row, 8 waves per block
// ---------------------------------------------------------------------------
__global__ __launch_bounds__(256)
void prep_context_kernel(const float* __restrict__ ctx, const float* __restrict__ w0,
                         unsigned short* __restrict__ cwm, float* __restrict__ sc) {
  const int row  = blockIdx.x * 8 + (threadIdx.x >> 5);
  const int lane = threadIdx.x & 31;
  const float* crow = ctx + (size_t)row * DD;
  unsigned short* orow = cwm + (size_t)row * DD;
  const float* wc = w0;            // w0[0:D]
  const float* wm = w0 + 2 * DD;   // w0[2D:3D]
  float acc = 0.f;
#pragma unroll
  for (int j = 0; j < 8; ++j) {
    int k = lane + 32 * j;
    float v = crow[k];
    acc += v * wc[k];
    orow[k] = f2bf(v * wm[k]);
  }
#pragma unroll
  for (int off = 16; off >= 1; off >>= 1) acc += __shfl_xor(acc, off, 32);
  if (lane == 0) sc[row] = acc;
}

__global__ __launch_bounds__(256)
void prep_query_kernel(const float* __restrict__ qry, const float* __restrict__ w0,
                       unsigned short* __restrict__ qbf, float* __restrict__ sq) {
  const int row  = blockIdx.x * 8 + (threadIdx.x >> 5);
  const int lane = threadIdx.x & 31;
  const float* qrow = qry + (size_t)row * DD;
  unsigned short* orow = qbf + (size_t)row * DD;
  const float* wq = w0 + DD;       // w0[D:2D]
  float acc = 0.f;
#pragma unroll
  for (int j = 0; j < 8; ++j) {
    int k = lane + 32 * j;
    float v = qrow[k];
    acc += v * wq[k];
    orow[k] = f2bf(v);
  }
#pragma unroll
  for (int off = 16; off >= 1; off >>= 1) acc += __shfl_xor(acc, off, 32);
  if (lane == 0) sq[row] = acc;
}

// ---------------------------------------------------------------------------
// f32 (batch,R,C) -> bf16 (batch,C,R) via LDS 32x32 tiles
// ---------------------------------------------------------------------------
template<int R, int C>
__global__ __launch_bounds__(256)
void transpose_bf16_kernel(const float* __restrict__ in, unsigned short* __restrict__ out) {
  __shared__ unsigned short tile[32][33];
  const int b  = blockIdx.z;
  const int c0 = blockIdx.x * 32;
  const int r0 = blockIdx.y * 32;
  const float* inb = in + (size_t)b * R * C;
  unsigned short* outb = out + (size_t)b * R * C;
#pragma unroll
  for (int i = 0; i < 4; ++i) {
    int r = r0 + threadIdx.y + i * 8;
    tile[threadIdx.y + i * 8][threadIdx.x] = f2bf(inb[(size_t)r * C + c0 + threadIdx.x]);
  }
  __syncthreads();
#pragma unroll
  for (int i = 0; i < 4; ++i) {
    int c = c0 + threadIdx.y + i * 8;
    outb[(size_t)c * R + r0 + threadIdx.x] = tile[threadIdx.x][threadIdx.y + i * 8];
  }
}

// ---------------------------------------------------------------------------
// Generic NT GEMM, bf16 in / f32 accumulate, N == 256 always.
//   out[m][n] = sum_k A[m][k] * BT[n][k]
// One wave computes a 16(m) x 256(n) strip: A-fragment loaded once per K-step,
// reused across all 16 n-tiles (16 x v8f accumulators = 128 VGPRs).
// Epilogues: SIM (bias+mask, f32), TR (bf16 transposed store), OUT (f32).
// ---------------------------------------------------------------------------
enum { EPI_SIM = 0, EPI_TR = 1, EPI_OUT = 2 };

template<int EPI>
__global__ __launch_bounds__(256)
void gemm_nt_kernel(const unsigned short* __restrict__ A,
                    const unsigned short* __restrict__ BT,
                    int M, int K,
                    float* __restrict__ fout, size_t out_bstride,
                    unsigned short* __restrict__ trout,
                    const float* __restrict__ sc, const float* __restrict__ sq,
                    const unsigned char* __restrict__ cmask,
                    const unsigned char* __restrict__ qmask) {
  const int b    = blockIdx.y;
  const int wave = threadIdx.x >> 5;
  const int lane = threadIdx.x & 31;
  const int m0   = (blockIdx.x * 8 + wave) * 16;
  if (m0 >= M) return;                      // wave-uniform: EXEC stays all-ones

  const unsigned short* Ab = A  + (size_t)b * M * (size_t)K;
  const unsigned short* Bb = BT + (size_t)b * 256 * (size_t)K;
  const int hi  = lane >> 4;                // lane half (0/1)
  const int l15 = lane & 15;

  // A fragment: row m0+l15, K split per ISA 16x32 bf16 layout
  const unsigned short* arow = Ab + (size_t)(m0 + l15) * K + hi * 8;
  // B fragment: column n = nt*16 + l15, K split per ISA 16x32 bf16 layout
  const unsigned short* bcol = Bb + (size_t)l15 * K + hi * 8;

  v8f acc[16] = {};

  for (int kb = 0; kb < K; kb += 32) {
    __builtin_prefetch(arow + kb + 128, 0, 1);   // global_prefetch_b8
    u16x8 alo = *(const u16x8*)(arow + kb);
    u16x8 ahi = *(const u16x8*)(arow + kb + 16);
    v16bf af = __builtin_bit_cast(
        v16bf, __builtin_shufflevector(alo, ahi, 0, 1, 2, 3, 4, 5, 6, 7,
                                       8, 9, 10, 11, 12, 13, 14, 15));
#pragma unroll
    for (int nt = 0; nt < 16; ++nt) {
      const unsigned short* bp = bcol + (size_t)nt * 16 * K + kb;
      u16x8 blo = *(const u16x8*)(bp);
      u16x8 bhi = *(const u16x8*)(bp + 16);
      v16bf bf = __builtin_bit_cast(
          v16bf, __builtin_shufflevector(blo, bhi, 0, 1, 2, 3, 4, 5, 6, 7,
                                         8, 9, 10, 11, 12, 13, 14, 15));
      acc[nt] = __builtin_amdgcn_wmma_f32_16x16x32_bf16(
          false, af, false, bf, (short)0, acc[nt], false, false);
    }
  }

  const int mbase = m0 + hi * 8;             // D-matrix layout: M = v + hi*8
#pragma unroll
  for (int nt = 0; nt < 16; ++nt) {
    const int n = nt * 16 + l15;
    if (EPI == EPI_SIM) {
      float sqv = sq[b * 256 + n];
      bool  qm  = qmask[b * 256 + n] != 0;
      float* ob = fout + (size_t)b * out_bstride;
#pragma unroll
      for (int v = 0; v < 8; ++v) {
        int m = mbase + v;
        float val = acc[nt][v] + sc[(size_t)b * M + m] + sqv;
        if (qm || (cmask[(size_t)b * M + m] != 0)) val = -1e30f;
        ob[(size_t)m * 256 + n] = val;
      }
    } else if (EPI == EPI_OUT) {
      float* ob = fout + (size_t)b * out_bstride;
#pragma unroll
      for (int v = 0; v < 8; ++v)
        ob[(size_t)(mbase + v) * 256 + n] = acc[nt][v];
    } else {                                 // EPI_TR: trout[b][n][m] bf16
      u16x8 pk;
#pragma unroll
      for (int v = 0; v < 8; ++v) pk[v] = f2bf(acc[nt][v]);
      *(u16x8*)(trout + (size_t)b * 256 * M + (size_t)n * M + mbase) = pk;
    }
  }
}

// ---------------------------------------------------------------------------
// Column softmax (over c axis): SStr[b][q][c] = softmax_c(sim[b][c][q]), bf16
// Block handles 16 q-columns over all 2048 c.
// ---------------------------------------------------------------------------
__global__ __launch_bounds__(256)
void softmax_col_kernel(const float* __restrict__ sim, unsigned short* __restrict__ sst) {
  __shared__ float red[16][17];
  __shared__ float cmax[16];
  __shared__ float cinv[16];
  const int b  = blockIdx.y;
  const int q0 = blockIdx.x * 16;
  const float* sb = sim + (size_t)b * LC * LQ;
  const int t = threadIdx.x;
  const int qi = t & 15, cr = t >> 4;

  float m = -3.0e38f;
  for (int c = cr; c < LC; c += 16) m = fmaxf(m, sb[(size_t)c * LQ + q0 + qi]);
  red[cr][qi] = m;
  __syncthreads();
  if (t < 16) {
    float mm = red[0][t];
#pragma unroll
    for (int i = 1; i < 16; ++i) mm = fmaxf(mm, red[i][t]);
    cmax[t] = mm;
  }
  __syncthreads();
  const float mx = cmax[qi];
  float s = 0.f;
  for (int c = cr; c < LC; c += 16) s += __expf(sb[(size_t)c * LQ + q0 + qi] - mx);
  red[cr][qi] = s;
  __syncthreads();
  if (t < 16) {
    float ss = 0.f;
#pragma unroll
    for (int i = 0; i < 16; ++i) ss += red[i][t];
    cinv[t] = 1.f / ss;
  }
  __syncthreads();
  const int qo = t >> 4, ci = t & 15;        // coalesced transposed writes
  const float mx2 = cmax[qo], rs = cinv[qo];
  unsigned short* orow = sst + (size_t)b * LQ * LC + (size_t)(q0 + qo) * LC;
  for (int c = ci; c < LC; c += 16)
    orow[c] = f2bf(__expf(sb[(size_t)c * LQ + q0 + qo] - mx2) * rs);
}

// ---------------------------------------------------------------------------
// Row softmax (over q axis): Sbf[b][c][q] bf16; one wave per row.
// ---------------------------------------------------------------------------
__global__ __launch_bounds__(256)
void softmax_row_kernel(const float* __restrict__ sim, unsigned short* __restrict__ sbf) {
  const int row  = blockIdx.x * 8 + (threadIdx.x >> 5);
  const int lane = threadIdx.x & 31;
  const float* r = sim + (size_t)row * LQ;
  float v[8];
  float m = -3.0e38f;
#pragma unroll
  for (int j = 0; j < 8; ++j) { v[j] = r[lane + 32 * j]; m = fmaxf(m, v[j]); }
#pragma unroll
  for (int off = 16; off >= 1; off >>= 1) m = fmaxf(m, __shfl_xor(m, off, 32));
  float s = 0.f;
#pragma unroll
  for (int j = 0; j < 8; ++j) { v[j] = __expf(v[j] - m); s += v[j]; }
#pragma unroll
  for (int off = 16; off >= 1; off >>= 1) s += __shfl_xor(s, off, 32);
  const float inv = 1.f / s;
  unsigned short* o = sbf + (size_t)row * LQ;
#pragma unroll
  for (int j = 0; j < 8; ++j) o[lane + 32 * j] = f2bf(v[j] * inv);
}

// ---------------------------------------------------------------------------
extern "C" void kernel_launch(void* const* d_in, const int* in_sizes, int n_in,
                              void* d_out, int out_size, void* d_ws, size_t ws_size,
                              hipStream_t stream) {
  const float*         context = (const float*)d_in[0];
  const float*         query   = (const float*)d_in[1];
  const unsigned char* cmask   = (const unsigned char*)d_in[2];
  const unsigned char* qmask   = (const unsigned char*)d_in[3];
  const float*         w0      = (const float*)d_in[4];
  float*               out     = (float*)d_out;

  char* ws = (char*)d_ws;
  size_t off = 0;
  auto carve = [&](size_t bytes) -> void* {
    void* p = ws + off;
    off += (bytes + 255) & ~(size_t)255;
    return p;
  };

  float*          sim  = (float*)carve((size_t)NB * LC * LQ * 4);
  unsigned short* cwm  = (unsigned short*)carve((size_t)NB * LC * DD * 2);
  unsigned short* cbfT = (unsigned short*)carve((size_t)NB * LC * DD * 2);
  unsigned short* qbf  = (unsigned short*)carve((size_t)NB * LQ * DD * 2);
  unsigned short* qbfT = (unsigned short*)carve((size_t)NB * LQ * DD * 2);
  unsigned short* SStr = (unsigned short*)carve((size_t)NB * LQ * LC * 2);
  unsigned short* Sbf  = (unsigned short*)carve((size_t)NB * LC * LQ * 2);
  unsigned short* Ttr  = (unsigned short*)carve((size_t)NB * DD * LQ * 2);
  float*          sc   = (float*)carve((size_t)NB * LC * 4);
  float*          sq   = (float*)carve((size_t)NB * LQ * 4);

  prep_context_kernel<<<NB * LC / 8, 256, 0, stream>>>(context, w0, cwm, sc);
  prep_query_kernel<<<NB * LQ / 8, 256, 0, stream>>>(query, w0, qbf, sq);
  transpose_bf16_kernel<LC, DD><<<dim3(DD / 32, LC / 32, NB), dim3(32, 8), 0, stream>>>(context, cbfT);
  transpose_bf16_kernel<LQ, DD><<<dim3(DD / 32, LQ / 32, NB), dim3(32, 8), 0, stream>>>(query, qbfT);

  // sim = cwm . qbf^T + sc + sq, masked  (M=LC, K=D)
  gemm_nt_kernel<EPI_SIM><<<dim3(LC / 128, NB), 256, 0, stream>>>(
      cwm, qbf, LC, DD, sim, (size_t)LC * LQ, nullptr, sc, sq, cmask, qmask);

  softmax_col_kernel<<<dim3(LQ / 16, NB), 256, 0, stream>>>(sim, SStr);
  softmax_row_kernel<<<NB * LC / 8, 256, 0, stream>>>(sim, Sbf);

  // Ttr = (SStr . cbfT^T)^T  (M=LQ, K=LC), stored bf16 transposed
  gemm_nt_kernel<EPI_TR><<<dim3(LQ / 128, NB), 256, 0, stream>>>(
      SStr, cbfT, LQ, LC, nullptr, 0, Ttr, nullptr, nullptr, nullptr, nullptr);

  // A = Sbf . qbfT^T -> out[0 : B*LC*D]  (M=LC, K=LQ)
  gemm_nt_kernel<EPI_OUT><<<dim3(LC / 128, NB), 256, 0, stream>>>(
      Sbf, qbfT, LC, LQ, out, (size_t)LC * DD, nullptr, nullptr, nullptr, nullptr, nullptr);

  // Bmat = Sbf . Ttr^T -> out[B*LC*D : 2*B*LC*D]
  gemm_nt_kernel<EPI_OUT><<<dim3(LC / 128, NB), 256, 0, stream>>>(
      Sbf, Ttr, LC, LQ, out + (size_t)NB * LC * DD, (size_t)LC * DD,
      nullptr, nullptr, nullptr, nullptr, nullptr);
}
